// MultiHeadAttention_76862734729513
// MI455X (gfx1250) — matrix operand order repaired and checked by reading, not compiled, not run on hardware
//
#include <hip/hip_runtime.h>
#include <hip/hip_bf16.h>
#include <cstddef>

// ---------------------------------------------------------------------------
// DeepViT re-attention MHA for MI455X (gfx1250, wave32, WMMA).
// All GEMMs run through v_wmma_f32_16x16x32_bf16 (fp32 accumulate).
// GEMM staging uses CDNA5 async global->LDS DMA (ASYNCcnt) with
// double-buffered LDS so stage k+1 fetch overlaps stage k WMMA.
// Workspace layout (bf16), ~192MB total:
//   Xq/Xk/Xv   [B*S, D]          3 x 8MB   (inputs transposed to (B,S,D))
//   Wq/Wk/Wv/Wo[D, D]            4 x 2MB
//   Qh/Kh      [B*H, S, DK]      2 x 8MB   (head-split, K-contiguous)
//   Vt         [B*H, DK, S]      8MB       (transposed: B-operand for attn@V)
//   Y1         [B*S, D]          8MB       (merged heads)
//   attn       [B*H, S, S]       128MB     (softmax probs, then attn2 in-place)
// ---------------------------------------------------------------------------

#define S_LEN 1024
#define NB 4
#define DM 1024
#define NH 16
#define DKH 64

typedef __attribute__((ext_vector_type(16))) __bf16 bf16x16;
typedef __attribute__((ext_vector_type(8))) float v8f;
typedef int v4i __attribute__((vector_size(16)));

#if defined(__HIP_DEVICE_COMPILE__) && defined(__gfx1250__) && \
    __has_builtin(__builtin_amdgcn_global_load_async_to_lds_b128)
#define HAVE_ASYNC_LDS 1
#else
#define HAVE_ASYNC_LDS 0
#endif

__device__ __forceinline__ unsigned short f2bf(float f) {
  unsigned int u = __float_as_uint(f);
  return (unsigned short)((u + 0x7FFFu + ((u >> 16) & 1u)) >> 16);
}
__device__ __forceinline__ float bf2f(unsigned short h) {
  return __uint_as_float(((unsigned int)h) << 16);
}

// 16B global -> LDS copy. Async DMA (no VGPR round trip) when available.
// Builtin signature (from probe diagnostic): (AS1 v4i*, AS3 v4i*, Ii, Ii).
__device__ __forceinline__ void copy16(unsigned short* l, const unsigned short* g) {
#if HAVE_ASYNC_LDS
  __builtin_amdgcn_global_load_async_to_lds_b128(
      (__attribute__((address_space(1))) v4i*)g,
      (__attribute__((address_space(3))) v4i*)l, 0, 0);
#else
  *(uint4*)l = *(const uint4*)g;
#endif
}

__device__ __forceinline__ void wait_async_copies() {
#if HAVE_ASYNC_LDS
#if __has_builtin(__builtin_amdgcn_s_wait_asynccnt)
  __builtin_amdgcn_s_wait_asynccnt(0);
#else
  asm volatile("s_wait_asynccnt 0x0" ::: "memory");
#endif
#endif
}

union ABFrag { bf16x16 v; uint4 q[2]; };

// A fragment (16x32 bf16, M x K): lane m=lane&15; lanes<16 hold K 0-7 & 16-23,
// lanes>=16 hold K 8-15 & 24-31 (ISA 7.12.2). Rows K-contiguous in memory.
__device__ __forceinline__ bf16x16 load_a_frag(const unsigned short* p, int ld) {
  int lane = threadIdx.x & 31;
  int m = lane & 15;
  int klo = (lane >> 4) << 3;
  ABFrag f;
  f.q[0] = *(const uint4*)(p + (size_t)m * ld + klo);
  f.q[1] = *(const uint4*)(p + (size_t)m * ld + klo + 16);
  return f.v;
}
// B fragment (32x16 bf16, K x N) where memory holds Bt[n][k] (K-contiguous
// rows, i.e. W rows): lane n=lane&15; lanes<16 hold K 0-7 & 16-23, lanes>=16 K 8-15 & 24-31.
__device__ __forceinline__ bf16x16 load_b_frag(const unsigned short* p, int ld) {
  int lane = threadIdx.x & 31;
  int n = lane & 15;
  int kb = (lane >> 4) << 3;
  ABFrag f;
  f.q[0] = *(const uint4*)(p + (size_t)n * ld + kb);
  f.q[1] = *(const uint4*)(p + (size_t)n * ld + kb + 16);
  return f.v;
}

__device__ __forceinline__ v8f wmma_bf16(bf16x16 a, bf16x16 b, v8f c) {
  return __builtin_amdgcn_wmma_f32_16x16x32_bf16(false, a, false, b,
                                                 (short)0, c, false, false);
}

// ---------------------------------------------------------------------------
// Generic 128x64 output tile GEMM core: C = A @ Bt^T, A:[128][lda], Bt:[64][ldb]
// (both bf16, K-contiguous rows, pointers pre-offset to tile origin).
// Block = 256 threads = 8 waves, wave grid 4(M) x 2(N), 32x32 per wave.
// Double-buffered LDS; async DMA prefetch of stage i+1 overlaps stage i WMMA.
// ---------------------------------------------------------------------------
#define LDS_STRIDE 40  // 32 halves + 8 pad; rows stay 16B aligned (80B)

__device__ __forceinline__ void gemm_tile_128x64(
    const unsigned short* __restrict__ A, int lda,
    const unsigned short* __restrict__ Bt, int ldb,
    int K, v8f c[2][2]) {
  __shared__ unsigned short sA[2][128 * LDS_STRIDE];
  __shared__ unsigned short sB[2][64 * LDS_STRIDE];
  const int tid = threadIdx.x;
  const int wid = tid >> 5;
  const int wm = wid & 3, wn = wid >> 2;

  // Per-thread staging assignment: 16B chunks. A tile = 512 chunks (2/thread),
  // B tile = 256 chunks (1/thread). Same (row, chunk) every stage.
  const int crow = tid >> 2;
  const int ccol = (tid & 3) * 8;  // halves
  const unsigned short* gA0 = A + (size_t)crow * lda + ccol;
  const unsigned short* gA1 = A + (size_t)(crow + 64) * lda + ccol;
  const unsigned short* gB = Bt + (size_t)crow * ldb + ccol;
  const int lofsA0 = crow * LDS_STRIDE + ccol;
  const int lofsA1 = (crow + 64) * LDS_STRIDE + ccol;
  const int lofsB = crow * LDS_STRIDE + ccol;

  const v8f zero = {0.f, 0.f, 0.f, 0.f, 0.f, 0.f, 0.f, 0.f};
#pragma unroll
  for (int i = 0; i < 2; ++i)
#pragma unroll
    for (int j = 0; j < 2; ++j) c[i][j] = zero;

  // Prologue: stage 0 into buffer 0.
  copy16(&sA[0][lofsA0], gA0);
  copy16(&sA[0][lofsA1], gA1);
  copy16(&sB[0][lofsB], gB);

  const int nst = K >> 5;
  for (int s = 0; s < nst; ++s) {
    const int buf = s & 1;
    wait_async_copies();  // own stage-s chunks landed in LDS
    __syncthreads();      // whole tile visible; prev buffer fully consumed
    if (s + 1 < nst) {    // overlap next-stage DMA with this stage's math
      const int k1 = (s + 1) << 5;
      copy16(&sA[buf ^ 1][lofsA0], gA0 + k1);
      copy16(&sA[buf ^ 1][lofsA1], gA1 + k1);
      copy16(&sB[buf ^ 1][lofsB], gB + k1);
    }
    const unsigned short* aw = &sA[buf][(wm * 32) * LDS_STRIDE];
    const unsigned short* bw = &sB[buf][(wn * 32) * LDS_STRIDE];
    bf16x16 a0 = load_a_frag(aw, LDS_STRIDE);
    bf16x16 a1 = load_a_frag(aw + 16 * LDS_STRIDE, LDS_STRIDE);
    bf16x16 b0 = load_b_frag(bw, LDS_STRIDE);
    bf16x16 b1 = load_b_frag(bw + 16 * LDS_STRIDE, LDS_STRIDE);
    c[0][0] = wmma_bf16(a0, b0, c[0][0]);
    c[0][1] = wmma_bf16(a0, b1, c[0][1]);
    c[1][0] = wmma_bf16(a1, b0, c[1][0]);
    c[1][1] = wmma_bf16(a1, b1, c[1][1]);
  }
}

// ---------------------------------------------------------------------------
// K1a: convert (S,B,D) fp32 -> (B*S, D) bf16, 4 elements/thread.
// ---------------------------------------------------------------------------
__global__ __launch_bounds__(256) void pack_x_kernel(
    const float* __restrict__ x, unsigned short* __restrict__ out) {
  size_t i = ((size_t)blockIdx.x * 256 + threadIdx.x) * 4;  // s*4096+b*1024+d
  int d = (int)(i & 1023);
  int b = (int)((i >> 10) & 3);
  int s = (int)(i >> 12);
  float4 f = *(const float4*)(x + i);
  uint2 o;
  o.x = (unsigned)f2bf(f.x) | ((unsigned)f2bf(f.y) << 16);
  o.y = (unsigned)f2bf(f.z) | ((unsigned)f2bf(f.w) << 16);
  *(uint2*)(out + (((size_t)b << 10) + s) * DM + d) = o;
}

// K1b: straight fp32 -> bf16 copy (weights), 4 elements/thread.
__global__ __launch_bounds__(256) void pack_w_kernel(
    const float* __restrict__ w, unsigned short* __restrict__ out) {
  size_t i = ((size_t)blockIdx.x * 256 + threadIdx.x) * 4;
  float4 f = *(const float4*)(w + i);
  uint2 o;
  o.x = (unsigned)f2bf(f.x) | ((unsigned)f2bf(f.y) << 16);
  o.y = (unsigned)f2bf(f.z) | ((unsigned)f2bf(f.w) << 16);
  *(uint2*)(out + i) = o;
}

// ---------------------------------------------------------------------------
// K2: projection GEMM  out = X @ W^T + bias, head-split output.
// vmode 0: out[((b*H+h)*S + s)*DK + dk]   (Q/K layout)
// vmode 1: out[((b*H+h)*DK + dk)*S + s]   (V transposed layout)
// grid (32, 16), block 256
// ---------------------------------------------------------------------------
__global__ __launch_bounds__(256) void proj_gemm_kernel(
    const unsigned short* __restrict__ X, const unsigned short* __restrict__ W,
    const float* __restrict__ bias, unsigned short* __restrict__ out,
    int vmode) {
  v8f c[2][2];
  const unsigned short* At = X + (size_t)blockIdx.x * 128 * DM;
  const unsigned short* Bt = W + (size_t)blockIdx.y * 64 * DM;
  gemm_tile_128x64(At, DM, Bt, DM, DM, c);

  const int lane = threadIdx.x & 31;
  const int wid = threadIdx.x >> 5;
  const int wm = wid & 3, wn = wid >> 2;
#pragma unroll
  for (int i = 0; i < 2; ++i)
#pragma unroll
    for (int j = 0; j < 2; ++j)
#pragma unroll
      for (int r = 0; r < 8; ++r) {
        int ml = wm * 32 + i * 16 + ((lane >> 4) << 3) + r;
        int nl = wn * 32 + j * 16 + (lane & 15);
        int gm = blockIdx.x * 128 + ml;  // b*S + s
        int gn = blockIdx.y * 64 + nl;   // h*DK + dk
        float v = c[i][j][r] + bias[gn];
        int b = gm >> 10, s = gm & 1023;
        int h = gn >> 6, dk = gn & 63;
        size_t o = vmode ? (((size_t)((b * NH + h) * DKH + dk)) << 10) + s
                         : (((size_t)((b * NH + h) * S_LEN + s)) << 6) + dk;
        out[o] = f2bf(v);
      }
}

// ---------------------------------------------------------------------------
// K3: scores = (Q @ K^T) / 8, row softmax, probs -> attn (bf16).
// grid (S/16, B*H); block 256 (8 waves). Each wave: 16 q-rows x 128 k-cols.
// Softmax kept in registers; cross-wave reduce via tiny LDS.
// ---------------------------------------------------------------------------
__global__ __launch_bounds__(256) void attn_softmax_kernel(
    const unsigned short* __restrict__ Qh, const unsigned short* __restrict__ Kh,
    unsigned short* __restrict__ attn) {
  __shared__ float red[16][8];  // [row][wave]
  const int bh = blockIdx.y;
  const int qb = blockIdx.x * 16;
  const int tid = threadIdx.x, lane = tid & 31, wave = tid >> 5;

  const unsigned short* Qbase = Qh + ((size_t)bh << 16) + (size_t)qb * DKH;
  const unsigned short* Kbase = Kh + ((size_t)bh << 16) + (size_t)(wave * 128) * DKH;

  const v8f zero = {0.f, 0.f, 0.f, 0.f, 0.f, 0.f, 0.f, 0.f};
  v8f c[8];
#pragma unroll
  for (int j = 0; j < 8; ++j) c[j] = zero;

#pragma unroll
  for (int ks = 0; ks < DKH; ks += 32) {
    bf16x16 a = load_a_frag(Qbase + ks, DKH);
#pragma unroll
    for (int j = 0; j < 8; ++j) {
      bf16x16 b = load_b_frag(Kbase + (size_t)(j * 16) * DKH + ks, DKH);
      c[j] = wmma_bf16(a, b, c[j]);
    }
  }
  // scale 1/sqrt(DK)
#pragma unroll
  for (int j = 0; j < 8; ++j)
#pragma unroll
    for (int r = 0; r < 8; ++r) c[j][r] *= 0.125f;

  const int rowbase = (lane >> 4) << 3;  // lanes<16: rows 0-7, lanes>=16: 8-15

  // ---- row max ----
  float vm[8];
#pragma unroll
  for (int r = 0; r < 8; ++r) {
    float m = c[0][r];
#pragma unroll
    for (int j = 1; j < 8; ++j) m = fmaxf(m, c[j][r]);
#pragma unroll
    for (int off = 1; off < 16; off <<= 1) m = fmaxf(m, __shfl_xor(m, off, 32));
    vm[r] = m;
  }
  if (lane == 0) {
#pragma unroll
    for (int r = 0; r < 8; ++r) red[r][wave] = vm[r];
  }
  if (lane == 16) {
#pragma unroll
    for (int r = 0; r < 8; ++r) red[8 + r][wave] = vm[r];
  }
  __syncthreads();
  float gmax[8];
#pragma unroll
  for (int r = 0; r < 8; ++r) {
    float m = -1e30f;
#pragma unroll
    for (int w = 0; w < 8; ++w) m = fmaxf(m, red[rowbase + r][w]);
    gmax[r] = m;
  }
  __syncthreads();

  // ---- exp + row sum ----
  float vs[8];
#pragma unroll
  for (int r = 0; r < 8; ++r) {
    float ssum = 0.f;
#pragma unroll
    for (int j = 0; j < 8; ++j) {
      float e = __expf(c[j][r] - gmax[r]);
      c[j][r] = e;
      ssum += e;
    }
#pragma unroll
    for (int off = 1; off < 16; off <<= 1) ssum += __shfl_xor(ssum, off, 32);
    vs[r] = ssum;
  }
  if (lane == 0) {
#pragma unroll
    for (int r = 0; r < 8; ++r) red[r][wave] = vs[r];
  }
  if (lane == 16) {
#pragma unroll
    for (int r = 0; r < 8; ++r) red[8 + r][wave] = vs[r];
  }
  __syncthreads();
  float inv[8];
#pragma unroll
  for (int r = 0; r < 8; ++r) {
    float d = 0.f;
#pragma unroll
    for (int w = 0; w < 8; ++w) d += red[rowbase + r][w];
    inv[r] = 1.f / d;
  }

  // ---- store probs bf16 ----
#pragma unroll
  for (int j = 0; j < 8; ++j)
#pragma unroll
    for (int r = 0; r < 8; ++r) {
      int row = rowbase + r;
      int col = wave * 128 + j * 16 + (lane & 15);
      size_t o = (((size_t)(bh * S_LEN + qb + row)) << 10) + col;
      attn[o] = f2bf(c[j][r] * inv[r]);
    }
}

// ---------------------------------------------------------------------------
// K4: DeepViT re-attention: mix heads with head_att, LayerNorm over head dim,
// write attn2 in place (bf16) and att_map (fp32). One thread per (b,q,k).
// ---------------------------------------------------------------------------
__global__ __launch_bounds__(256) void reattn_kernel(
    unsigned short* __restrict__ attn, const float* __restrict__ head_att,
    const float* __restrict__ ln_g, const float* __restrict__ ln_b,
    float* __restrict__ att_map) {
  __shared__ float sha[256];
  __shared__ float sg[16], sb[16];
  const int tid = threadIdx.x;
  sha[tid] = head_att[tid];
  if (tid < 16) { sg[tid] = ln_g[tid]; sb[tid] = ln_b[tid]; }
  __syncthreads();

  size_t idx = (size_t)blockIdx.x * 256 + tid;  // b*S*S + q*S + k
  int b = (int)(idx >> 20);
  int q = (int)((idx >> 10) & 1023);
  int k = (int)(idx & 1023);
  const size_t hstride = (size_t)S_LEN * S_LEN;  // 1M
  size_t base = (size_t)b * NH * hstride + (size_t)q * S_LEN + k;

  float a[16];
#pragma unroll
  for (int h = 0; h < 16; ++h) a[h] = bf2f(attn[base + h * hstride]);

  float mix[16];
#pragma unroll
  for (int g = 0; g < 16; ++g) {
    float s = 0.f;
#pragma unroll
    for (int h = 0; h < 16; ++h) s += a[h] * sha[h * 16 + g];
    mix[g] = s;
  }
  float mean = 0.f;
#pragma unroll
  for (int g = 0; g < 16; ++g) mean += mix[g];
  mean *= (1.f / 16.f);
  float var = 0.f;
#pragma unroll
  for (int g = 0; g < 16; ++g) { float d = mix[g] - mean; var += d * d; }
  var *= (1.f / 16.f);
  float rinv = rsqrtf(var + 1e-5f);

  float msum = 0.f;
#pragma unroll
  for (int g = 0; g < 16; ++g) {
    float o = (mix[g] - mean) * rinv * sg[g] + sb[g];
    msum += o;
    attn[base + g * hstride] = f2bf(o);
  }
  att_map[idx] = msum * (1.f / 16.f);
}

// ---------------------------------------------------------------------------
// K5: y = attn2 @ V  per (b,h). grid (8, 1, 64); z = b*H+h.
// ---------------------------------------------------------------------------
__global__ __launch_bounds__(256) void av_gemm_kernel(
    const unsigned short* __restrict__ attn2, const unsigned short* __restrict__ Vt,
    unsigned short* __restrict__ Y1) {
  const int bh = blockIdx.z;
  v8f c[2][2];
  const unsigned short* At =
      attn2 + ((size_t)bh << 20) + (size_t)blockIdx.x * 128 * S_LEN;
  const unsigned short* Bt = Vt + (size_t)bh * DKH * S_LEN;
  gemm_tile_128x64(At, S_LEN, Bt, S_LEN, S_LEN, c);

  const int lane = threadIdx.x & 31;
  const int wid = threadIdx.x >> 5;
  const int wm = wid & 3, wn = wid >> 2;
  const int b = bh >> 4, h = bh & 15;
#pragma unroll
  for (int i = 0; i < 2; ++i)
#pragma unroll
    for (int j = 0; j < 2; ++j)
#pragma unroll
      for (int r = 0; r < 8; ++r) {
        int ml = wm * 32 + i * 16 + ((lane >> 4) << 3) + r;
        int nl = wn * 32 + j * 16 + (lane & 15);  // dk in [0,64)
        int q = blockIdx.x * 128 + ml;
        size_t o = (((size_t)(b * S_LEN + q)) << 10) + (h * DKH + nl);
        Y1[o] = f2bf(c[i][j][r]);
      }
}

// ---------------------------------------------------------------------------
// K6: y_out = Y1 @ Wo^T + bo, stored transposed back to (S,B,D) fp32.
// grid (32, 16)
// ---------------------------------------------------------------------------
__global__ __launch_bounds__(256) void out_gemm_kernel(
    const unsigned short* __restrict__ Y1, const unsigned short* __restrict__ Wo,
    const float* __restrict__ bo, float* __restrict__ yout) {
  v8f c[2][2];
  const unsigned short* At = Y1 + (size_t)blockIdx.x * 128 * DM;
  const unsigned short* Bt = Wo + (size_t)blockIdx.y * 64 * DM;
  gemm_tile_128x64(At, DM, Bt, DM, DM, c);

  const int lane = threadIdx.x & 31;
  const int wid = threadIdx.x >> 5;
  const int wm = wid & 3, wn = wid >> 2;
#pragma unroll
  for (int i = 0; i < 2; ++i)
#pragma unroll
    for (int j = 0; j < 2; ++j)
#pragma unroll
      for (int r = 0; r < 8; ++r) {
        int ml = wm * 32 + i * 16 + ((lane >> 4) << 3) + r;
        int nl = wn * 32 + j * 16 + (lane & 15);
        int gm = blockIdx.x * 128 + ml;  // b*S + s
        int gn = blockIdx.y * 64 + nl;   // d
        int b = gm >> 10, s = gm & 1023;
        yout[(((size_t)(s * NB + b)) << 10) + gn] = c[i][j][r] + bo[gn];
      }
}

// ---------------------------------------------------------------------------
extern "C" void kernel_launch(void* const* d_in, const int* in_sizes, int n_in,
                              void* d_out, int out_size, void* d_ws,
                              size_t ws_size, hipStream_t stream) {
  (void)in_sizes; (void)n_in; (void)out_size; (void)ws_size;
  const float* query = (const float*)d_in[0];
  const float* key_i = (const float*)d_in[1];
  const float* value = (const float*)d_in[2];
  const float* Wq = (const float*)d_in[3];
  const float* bq = (const float*)d_in[4];
  const float* Wk = (const float*)d_in[5];
  const float* bk = (const float*)d_in[6];
  const float* Wv = (const float*)d_in[7];
  const float* bv = (const float*)d_in[8];
  const float* Wo = (const float*)d_in[9];
  const float* bo = (const float*)d_in[10];
  const float* head_att = (const float*)d_in[11];
  const float* ln_g = (const float*)d_in[12];
  const float* ln_b = (const float*)d_in[13];

  unsigned short* Xq = (unsigned short*)d_ws;
  unsigned short* Xk = Xq + 4194304;
  unsigned short* Xv = Xk + 4194304;
  unsigned short* Wqb = Xv + 4194304;
  unsigned short* Wkb = Wqb + 1048576;
  unsigned short* Wvb = Wkb + 1048576;
  unsigned short* Wob = Wvb + 1048576;
  unsigned short* Qh = Wob + 1048576;
  unsigned short* Kh = Qh + 4194304;
  unsigned short* Vt = Kh + 4194304;
  unsigned short* Y1 = Vt + 4194304;
  unsigned short* attn = Y1 + 4194304;  // 64M elements (128MB)

  float* yout = (float*)d_out;
  float* att_map = yout + 4194304;

  dim3 blk(256);
  pack_x_kernel<<<4096, blk, 0, stream>>>(query, Xq);
  pack_x_kernel<<<4096, blk, 0, stream>>>(key_i, Xk);
  pack_x_kernel<<<4096, blk, 0, stream>>>(value, Xv);
  pack_w_kernel<<<1024, blk, 0, stream>>>(Wq, Wqb);
  pack_w_kernel<<<1024, blk, 0, stream>>>(Wk, Wkb);
  pack_w_kernel<<<1024, blk, 0, stream>>>(Wv, Wvb);
  pack_w_kernel<<<1024, blk, 0, stream>>>(Wo, Wob);

  proj_gemm_kernel<<<dim3(32, 16), blk, 0, stream>>>(Xq, Wqb, bq, Qh, 0);
  proj_gemm_kernel<<<dim3(32, 16), blk, 0, stream>>>(Xk, Wkb, bk, Kh, 0);
  proj_gemm_kernel<<<dim3(32, 16), blk, 0, stream>>>(Xv, Wvb, bv, Vt, 1);

  attn_softmax_kernel<<<dim3(64, 64), blk, 0, stream>>>(Qh, Kh, attn);

  reattn_kernel<<<16384, blk, 0, stream>>>(attn, head_att, ln_g, ln_b, att_map);

  av_gemm_kernel<<<dim3(8, 1, 64), blk, 0, stream>>>(attn, Vt, Y1);

  out_gemm_kernel<<<dim3(32, 16), blk, 0, stream>>>(Y1, Wob, bo, yout);
}
